// QWenBlock_51866025066701
// MI455X (gfx1250) — matrix operand order repaired and checked by reading, not compiled, not run on hardware
//
#include <hip/hip_runtime.h>
#include <cstddef>
#include <cstdint>

// ---- problem dims (compile-time) ----
#define Bd  2
#define Sd  1024
#define Hd  4096
#define NHd 32
#define HDd 128
#define Id  11008
#define Md  (Bd * Sd)          // 2048
#define QKVLD (3 * Hd)         // 12288

typedef __attribute__((ext_vector_type(16))) __bf16 v16bf;
typedef __attribute__((ext_vector_type(8)))  __bf16 v8bf;
typedef __attribute__((ext_vector_type(8)))  float  v8f;
typedef __attribute__((ext_vector_type(4)))  unsigned tdm4;   // D# group0
typedef __attribute__((ext_vector_type(8)))  unsigned tdm8;   // D# group1

union FragBF { v16bf v; v8bf h[2]; };

__device__ __forceinline__ v8f wmma_bf16(v16bf a, v16bf b, v8f c) {
  // 8 args: (neg_a, A, neg_b, B, c_mod, C, reuse_a, reuse_b)
  return __builtin_amdgcn_wmma_f32_16x16x32_bf16(false, a, false, b, (short)0, c, false, false);
}

__device__ __forceinline__ void pack_store4(__bf16* p, float a, float b, float c, float d) {
  union { __bf16 h[4]; unsigned long long q; } u;
  u.h[0] = (__bf16)a; u.h[1] = (__bf16)b; u.h[2] = (__bf16)c; u.h[3] = (__bf16)d;
  *(unsigned long long*)p = u.q;
}

// ---------------- RMSNorm + cast to bf16 ----------------
__global__ __launch_bounds__(256) void rmsnorm_bf16_kernel(const float* __restrict__ x,
                                                           const float* __restrict__ w,
                                                           __bf16* __restrict__ out) {
  const int row = blockIdx.x;
  const int tid = threadIdx.x;
  const float* xr = x + (size_t)row * Hd;
  float ss = 0.f;
#pragma unroll 4
  for (int i = tid; i < Hd; i += 256) { float v = xr[i]; ss += v * v; }
#pragma unroll
  for (int off = 16; off > 0; off >>= 1) ss += __shfl_xor(ss, off, 32);
  __shared__ float red[8];
  __shared__ float sscale;
  if ((tid & 31) == 0) red[tid >> 5] = ss;
  __syncthreads();
  if (tid == 0) {
    float t = 0.f;
    for (int i = 0; i < 8; i++) t += red[i];
    sscale = rsqrtf(t / (float)Hd + 1e-6f);
  }
  __syncthreads();
  const float sc = sscale;
  for (int i = tid; i < Hd; i += 256)
    out[(size_t)row * Hd + i] = (__bf16)(xr[i] * sc * w[i]);
}

// ---------------- RoPE (in-place on bf16 q,k inside qkv) ----------------
__global__ __launch_bounds__(256) void rope_kernel(const int* __restrict__ positions,
                                                   __bf16* __restrict__ qkv) {
  const int idx = blockIdx.x * 256 + threadIdx.x;  // B*S*NH*64 threads
  const int d = idx & 63;
  const int h = (idx >> 6) & (NHd - 1);
  const int s = (idx >> 11) & (Sd - 1);
  const int b = idx >> 21;
  const float pos = (float)positions[s];
  const float inv = __powf(10000.f, -(float)d * (1.f / 64.f));
  const float f = pos * inv;
  const float cs = __cosf(f), sn = __sinf(f);
  const size_t base = (size_t)(b * Sd + s) * QKVLD + (size_t)h * HDd;
  __bf16* q = qkv + base;
  __bf16* k = qkv + base + Hd;
  const float q1 = (float)q[d], q2 = (float)q[d + 64];
  q[d]      = (__bf16)(q1 * cs - q2 * sn);
  q[d + 64] = (__bf16)(q2 * cs + q1 * sn);
  const float k1 = (float)k[d], k2 = (float)k[d + 64];
  k[d]      = (__bf16)(k1 * cs - k2 * sn);
  k[d + 64] = (__bf16)(k2 * cs + k1 * sn);
}

// ---------------- Tiled bf16 WMMA GEMM (A tile via TDM) ----------------
// C[M,N] = A[M,K](bf16) x W[K,N](f32, converted on the fly)
// EPI 0: out_bf16 = acc + bias[col]
// EPI 1: out_f32  = acc + res[idx]         (res may alias out; 1 touch/elem)
// EPI 2: out_bf16 = silu(accG) * accU      (DUAL weights W and W2)
template <int BN, int WM, int WN, int EPI>
__global__ __launch_bounds__(256) void gemm_kernel(const __bf16* __restrict__ A,
                                                   const float* __restrict__ W,
                                                   const float* __restrict__ W2,
                                                   const float* __restrict__ bias,
                                                   const float* __restrict__ res,
                                                   void* __restrict__ outp,
                                                   int K, int ldw, int ldout) {
  constexpr int  BM = 128;
  constexpr bool DUAL = (EPI == 2);
  constexpr int  WAVES_N = BN / (WN * 16);

  __shared__ __bf16 As[BM][32];                 // [m][k] (written by TDM)
  __shared__ __bf16 Bs[DUAL ? 2 : 1][BN][32];   // transposed [n][k]

  const int tid = threadIdx.x;
  const int lane = tid & 31;
  const int wave = tid >> 5;
  const int lan = lane & 15;
  const int halfid = lane >> 4;
  const int wm = wave / WAVES_N;
  const int wn = wave % WAVES_N;
  const int row0 = blockIdx.y * BM;
  const int col0 = blockIdx.x * BN;

  v8f acc[WM][WN];
  v8f acc2[DUAL ? WM : 1][DUAL ? WN : 1];
#pragma unroll
  for (int i = 0; i < WM; i++)
#pragma unroll
    for (int j = 0; j < WN; j++)
#pragma unroll
      for (int e = 0; e < 8; e++) {
        acc[i][j][e] = 0.f;
        if constexpr (DUAL) acc2[i][j][e] = 0.f;
      }

  // ---- TDM descriptor group1 (loop invariant): 2D tile 32(k) x 128(m), 2B elems
  const unsigned lds_a = (unsigned)(uintptr_t)&As[0][0];
  tdm8 g1;
  g1[0] = 0x00010000u;                                   // data_size = 1 (2 bytes)
  g1[1] = ((unsigned)K & 0xFFFFu) << 16;                 // tensor_dim0[15:0]
  g1[2] = ((unsigned)K >> 16) | ((unsigned)Md << 16);    // dim0[31:16] | dim1[15:0]
  g1[3] = (32u << 16);                                   // dim1[31:16]=0 | tile_dim0=32
  g1[4] = 128u;                                          // tile_dim1=128, tile_dim2=0
  g1[5] = (unsigned)K;                                   // tensor_dim0_stride[31:0]
  g1[6] = 0u;                                            // stride[47:32] | dim1_stride lo
  g1[7] = 0u;

  // W staging: each thread loads 4 k-rows x 4 n-cols, packs 4 bf16 along k,
  // stores 4x ds_store_b64 into the transposed tile.
  const int wk = (tid & 7) * 4;    // k0
  const int wns = tid >> 3;        // n-group of 4

  for (int kt = 0; kt < K; kt += 32) {
    if (wave == 0) {  // issue tensor DMA: A[row0:row0+128, kt:kt+32] -> As
      const unsigned long long ga =
          (unsigned long long)(uintptr_t)(A + (size_t)row0 * K + kt);
      tdm4 g0;
      g0[0] = 1u;                                            // count=1
      g0[1] = lds_a;                                         // lds_addr
      g0[2] = (unsigned)ga;                                  // global_addr[31:0]
      g0[3] = ((unsigned)(ga >> 32) & 0x01FFFFFFu) | 0x80000000u;  // addr[56:32]|type=2
      asm volatile("tensor_load_to_lds %0, %1" :: "s"(g0), "s"(g1) : "memory");
    }
    if (wns < BN / 4) {  // stage W tile(s): f32 -> bf16, transposed
      const float* wp = W + (size_t)(kt + wk) * ldw + col0 + wns * 4;
      const float4 r0 = *(const float4*)(wp);
      const float4 r1 = *(const float4*)(wp + ldw);
      const float4 r2 = *(const float4*)(wp + 2 * (size_t)ldw);
      const float4 r3 = *(const float4*)(wp + 3 * (size_t)ldw);
      pack_store4(&Bs[0][wns * 4 + 0][wk], r0.x, r1.x, r2.x, r3.x);
      pack_store4(&Bs[0][wns * 4 + 1][wk], r0.y, r1.y, r2.y, r3.y);
      pack_store4(&Bs[0][wns * 4 + 2][wk], r0.z, r1.z, r2.z, r3.z);
      pack_store4(&Bs[0][wns * 4 + 3][wk], r0.w, r1.w, r2.w, r3.w);
      if constexpr (DUAL) {
        const float* wp2 = W2 + (size_t)(kt + wk) * ldw + col0 + wns * 4;
        const float4 s0 = *(const float4*)(wp2);
        const float4 s1 = *(const float4*)(wp2 + ldw);
        const float4 s2 = *(const float4*)(wp2 + 2 * (size_t)ldw);
        const float4 s3 = *(const float4*)(wp2 + 3 * (size_t)ldw);
        pack_store4(&Bs[1][wns * 4 + 0][wk], s0.x, s1.x, s2.x, s3.x);
        pack_store4(&Bs[1][wns * 4 + 1][wk], s0.y, s1.y, s2.y, s3.y);
        pack_store4(&Bs[1][wns * 4 + 2][wk], s0.z, s1.z, s2.z, s3.z);
        pack_store4(&Bs[1][wns * 4 + 3][wk], s0.w, s1.w, s2.w, s3.w);
      }
    }
    if (wave == 0) __builtin_amdgcn_s_wait_tensorcnt(0);  // TDM done -> LDS visible
    __syncthreads();

    FragBF af[WM], bfr[WN], bfr2[DUAL ? WN : 1];
#pragma unroll
    for (int i = 0; i < WM; i++) {
      const int row = wm * (WM * 16) + i * 16 + lan;
      const int ka = halfid * 8;                  // A 16x32 bf16 layout
      af[i].h[0] = *(const v8bf*)&As[row][ka];
      af[i].h[1] = *(const v8bf*)&As[row][ka + 16];
    }
#pragma unroll
    for (int j = 0; j < WN; j++) {
      const int col = wn * (WN * 16) + j * 16 + lan;
      const int kb = halfid * 8;                  // B 32x16 bf16 layout
      bfr[j].h[0] = *(const v8bf*)&Bs[0][col][kb];
      bfr[j].h[1] = *(const v8bf*)&Bs[0][col][kb + 16];
      if constexpr (DUAL) {
        bfr2[j].h[0] = *(const v8bf*)&Bs[1][col][kb];
        bfr2[j].h[1] = *(const v8bf*)&Bs[1][col][kb + 16];
      }
    }
#pragma unroll
    for (int i = 0; i < WM; i++)
#pragma unroll
      for (int j = 0; j < WN; j++) {
        acc[i][j] = wmma_bf16(af[i].v, bfr[j].v, acc[i][j]);
        if constexpr (DUAL) acc2[i][j] = wmma_bf16(af[i].v, bfr2[j].v, acc2[i][j]);
      }
    __syncthreads();
  }

  // epilogue (C layout: VGPR r -> rows r / r+8 by lane half, cols = lane&15)
#pragma unroll
  for (int i = 0; i < WM; i++)
#pragma unroll
    for (int j = 0; j < WN; j++)
#pragma unroll
      for (int r = 0; r < 8; r++) {
        const int rowg = row0 + wm * (WM * 16) + i * 16 + r + 8 * halfid;
        const int colg = col0 + wn * (WN * 16) + j * 16 + lan;
        const size_t oidx = (size_t)rowg * ldout + colg;
        if constexpr (EPI == 0) {
          ((__bf16*)outp)[oidx] = (__bf16)(acc[i][j][r] + bias[colg]);
        } else if constexpr (EPI == 1) {
          ((float*)outp)[oidx] = acc[i][j][r] + res[oidx];
        } else {
          const float g = acc[i][j][r], u = acc2[i][j][r];
          const float sv = g / (1.f + __expf(-g));
          ((__bf16*)outp)[oidx] = (__bf16)(sv * u);
        }
      }
}

// ---------------- Flash attention (WMMA, causal, online softmax) ----------------
__global__ __launch_bounds__(256) void flash_attn_kernel(const __bf16* __restrict__ qkv,
                                                         __bf16* __restrict__ attn_out) {
  const int qb = blockIdx.x, h = blockIdx.y, b = blockIdx.z;
  const int tid = threadIdx.x, lane = tid & 31, w = tid >> 5;
  const int lan = lane & 15, halfid = lane >> 4;
  const int q0 = qb * 128;

  __shared__ __bf16 V_s[HDd][128];     // V transposed: [d][kv]
  __shared__ __bf16 P_s[8][16][128];   // per-wave P staging (rows x keys)

  // Q fragments straight from global (rows are contiguous bf16)
  FragBF qf[4];
  {
    const int srow = q0 + w * 16 + lan;
    const __bf16* qp = qkv + (size_t)(b * Sd + srow) * QKVLD + (size_t)h * HDd;
#pragma unroll
    for (int c = 0; c < 4; c++) {
      const int ka = c * 32 + halfid * 8;
      qf[c].h[0] = *(const v8bf*)(qp + ka);
      qf[c].h[1] = *(const v8bf*)(qp + ka + 16);
    }
  }

  v8f   O[8];
  float mrow[8], lrow[8];
#pragma unroll
  for (int dt = 0; dt < 8; dt++)
#pragma unroll
    for (int e = 0; e < 8; e++) O[dt][e] = 0.f;
#pragma unroll
  for (int r = 0; r < 8; r++) { mrow[r] = -3.0e38f; lrow[r] = 0.f; }

  for (int kb = 0; kb <= qb; kb++) {
    {  // stage V transposed into LDS (all 256 threads)
      const int kv = tid >> 1;
      const int dh = (tid & 1) * 64;
      const __bf16* vp = qkv + (size_t)(b * Sd + kb * 128 + kv) * QKVLD +
                         (size_t)(2 * Hd) + (size_t)h * HDd + dh;
#pragma unroll
      for (int c = 0; c < 8; c++) {
        v8bf t = *(const v8bf*)(vp + c * 8);
#pragma unroll
        for (int e = 0; e < 8; e++) V_s[dh + c * 8 + e][kv] = t[e];
      }
    }
    __syncthreads();

    // S = Q K^T  (K fragments straight from global; rows of K are the B columns)
    v8f S[8];
#pragma unroll
    for (int n = 0; n < 8; n++)
#pragma unroll
      for (int e = 0; e < 8; e++) S[n][e] = 0.f;
#pragma unroll
    for (int c = 0; c < 4; c++) {
#pragma unroll
      for (int n = 0; n < 8; n++) {
        const int key = kb * 128 + n * 16 + lan;
        const __bf16* kp = qkv + (size_t)(b * Sd + key) * QKVLD + Hd + (size_t)h * HDd;
        const int ka = c * 32 + halfid * 8;
        FragBF kf;
        kf.h[0] = *(const v8bf*)(kp + ka);
        kf.h[1] = *(const v8bf*)(kp + ka + 16);
        S[n] = wmma_bf16(qf[c].v, kf.v, S[n]);
      }
    }

    const bool diag = (kb == qb);
    float alpha[8];
#pragma unroll
    for (int r = 0; r < 8; r++) {
      const int qrow = q0 + w * 16 + r + 8 * halfid;
      float sv[8];
      float rmax = -3.0e38f;
#pragma unroll
      for (int n = 0; n < 8; n++) {
        float s = S[n][r] * 0.08838834764831845f;   // HD^-0.5
        if (diag) {
          const int kcol = kb * 128 + n * 16 + lan;
          if (kcol > qrow) s = -1.0e9f;
        }
        sv[n] = s;
        rmax = fmaxf(rmax, s);
      }
#pragma unroll
      for (int off = 8; off > 0; off >>= 1) rmax = fmaxf(rmax, __shfl_xor(rmax, off, 32));
      const float mnew = fmaxf(mrow[r], rmax);
      const float al = __expf(mrow[r] - mnew);
      float psum = 0.f;
#pragma unroll
      for (int n = 0; n < 8; n++) {
        const float p = __expf(sv[n] - mnew);
        psum += p;
        P_s[w][r + 8 * halfid][n * 16 + lan] = (__bf16)p;
      }
#pragma unroll
      for (int off = 8; off > 0; off >>= 1) psum += __shfl_xor(psum, off, 32);
      lrow[r] = lrow[r] * al + psum;
      mrow[r] = mnew;
      alpha[r] = al;
    }
#pragma unroll
    for (int dt = 0; dt < 8; dt++)
#pragma unroll
      for (int r = 0; r < 8; r++) O[dt][r] *= alpha[r];

    asm volatile("s_wait_dscnt 0" ::: "memory");  // wave-local P_s RAW

    // O += P V
#pragma unroll
    for (int ck = 0; ck < 4; ck++) {
      FragBF pf;
      const int ka = ck * 32 + halfid * 8;
      pf.h[0] = *(const v8bf*)&P_s[w][lan][ka];
      pf.h[1] = *(const v8bf*)&P_s[w][lan][ka + 16];
#pragma unroll
      for (int dt = 0; dt < 8; dt++) {
        FragBF vf;
        const int kvk = ck * 32 + halfid * 8;
        vf.h[0] = *(const v8bf*)&V_s[dt * 16 + lan][kvk];
        vf.h[1] = *(const v8bf*)&V_s[dt * 16 + lan][kvk + 16];
        O[dt] = wmma_bf16(pf.v, vf.v, O[dt]);
      }
    }
    __syncthreads();
  }

#pragma unroll
  for (int dt = 0; dt < 8; dt++)
#pragma unroll
    for (int r = 0; r < 8; r++) {
      const int qrow = q0 + w * 16 + r + 8 * halfid;
      const float v = O[dt][r] / lrow[r];
      attn_out[(size_t)(b * Sd + qrow) * Hd + (size_t)h * HDd + dt * 16 + lan] = (__bf16)v;
    }
}

// ---------------- workspace layout ----------------
static constexpr size_t O_XN  = 0;                                   // 2048*4096*2
static constexpr size_t O_QKV = O_XN  + (size_t)Md * Hd * 2;         // 2048*12288*2
static constexpr size_t O_AT  = O_QKV + (size_t)Md * QKVLD * 2;      // 2048*4096*2
static constexpr size_t O_XN2 = O_AT  + (size_t)Md * Hd * 2;         // 2048*4096*2
static constexpr size_t O_HH  = O_XN2 + (size_t)Md * Hd * 2;         // 2048*11008*2

extern "C" void kernel_launch(void* const* d_in, const int* in_sizes, int n_in,
                              void* d_out, int out_size, void* d_ws, size_t ws_size,
                              hipStream_t stream) {
  (void)in_sizes; (void)n_in; (void)out_size; (void)ws_size;
  const int*   positions = (const int*)d_in[0];
  const float* hidden    = (const float*)d_in[1];
  const float* ln1_w     = (const float*)d_in[2];
  const float* ln2_w     = (const float*)d_in[3];
  const float* w_qkv     = (const float*)d_in[4];
  const float* b_qkv     = (const float*)d_in[5];
  const float* w_o       = (const float*)d_in[6];
  const float* w_gu      = (const float*)d_in[7];
  const float* w_down    = (const float*)d_in[8];
  float* out = (float*)d_out;

  char* ws = (char*)d_ws;
  __bf16* xnorm  = (__bf16*)(ws + O_XN);
  __bf16* qkv    = (__bf16*)(ws + O_QKV);
  __bf16* attnb  = (__bf16*)(ws + O_AT);
  __bf16* xnorm2 = (__bf16*)(ws + O_XN2);
  __bf16* hbuf   = (__bf16*)(ws + O_HH);

  // 1) x = rmsnorm(hidden, ln1) -> bf16
  rmsnorm_bf16_kernel<<<Md, 256, 0, stream>>>(hidden, ln1_w, xnorm);

  // 2) qkv = x @ w_qkv + b_qkv   (M=2048, N=12288, K=4096) -> bf16
  gemm_kernel<128, 4, 2, 0><<<dim3(QKVLD / 128, Md / 128), 256, 0, stream>>>(
      xnorm, w_qkv, nullptr, b_qkv, nullptr, (void*)qkv, Hd, QKVLD, QKVLD);

  // 3) RoPE on q,k in place
  rope_kernel<<<(Bd * Sd * NHd * 64) / 256, 256, 0, stream>>>(positions, qkv);

  // 4) attention -> bf16 [M, H]
  flash_attn_kernel<<<dim3(Sd / 128, NHd, Bd), 256, 0, stream>>>(qkv, attnb);

  // 5) out = hidden + attn @ w_o  (f32)
  gemm_kernel<128, 4, 2, 1><<<dim3(Hd / 128, Md / 128), 256, 0, stream>>>(
      attnb, w_o, nullptr, nullptr, hidden, (void*)out, Hd, Hd, Hd);

  // 6) x2 = rmsnorm(out, ln2) -> bf16
  rmsnorm_bf16_kernel<<<Md, 256, 0, stream>>>(out, ln2_w, xnorm2);

  // 7) h = silu(x2 @ Wg) * (x2 @ Wu)  (fused dual-B GEMM, N=11008) -> bf16
  gemm_kernel<64, 2, 2, 2><<<dim3(Id / 64, Md / 128), 256, 0, stream>>>(
      xnorm2, w_gu, w_gu + Id, nullptr, nullptr, (void*)hbuf, Hd, 2 * Id, Id);

  // 8) out += h @ w_down  (res aliases out; each element written once)
  gemm_kernel<128, 4, 2, 1><<<dim3(Hd / 128, Md / 128), 256, 0, stream>>>(
      hbuf, w_down, nullptr, nullptr, out, (void*)out, Id, Hd, Hd);
}
